// StaticWindowedAttnBlock_7164005450356
// MI455X (gfx1250) — matrix operand order repaired and checked by reading, not compiled, not run on hardware
//
#include <hip/hip_runtime.h>

// ---------------------------------------------------------------------------
// CDNA5 (gfx1250) implementation of StaticWindowedAttnBlock
// D=8 L=2048 C=128 CP=16 H=4 HD=32 QB=32 KB=128 NQ=64  -> NTOK = 16384
// GEMMs via v_wmma_f32_16x16x32_f16; bias staged to LDS via async-LDS loads.
// ---------------------------------------------------------------------------

typedef __attribute__((ext_vector_type(16))) _Float16 v16h;
typedef __attribute__((ext_vector_type(8)))  _Float16 v8h;
typedef __attribute__((ext_vector_type(8)))  float    v8f;

#define NTOK  16384
#define CDIM  128
#define LSEQ  2048
#define NQBLK 64
#define QB_   32
#define KB_   128
#define NHEAD 4
#define HDIM  32

__device__ __forceinline__ float sigm(float x){ return 1.0f/(1.0f + __expf(-x)); }

__device__ __forceinline__ v8f zero8(){
  v8f z;
  #pragma unroll
  for(int i=0;i<8;i++) z[i]=0.0f;
  return z;
}

// A-fragment (16x32 f16, MxK): lane ml holds row m, lane-group g picks K halves
// [g*8, g*8+8) and [16+g*8, 16+g*8+8)   (ISA 7.12.2, 16-bit A 16x32 layout)
__device__ __forceinline__ v16h load_afrag(const _Float16* rowPtr, int g){
  v8h lo = *(const v8h*)(rowPtr + g*8);
  v8h hi = *(const v8h*)(rowPtr + 16 + g*8);
  v16h r;
  #pragma unroll
  for(int i=0;i<8;i++){ r[i]=lo[i]; r[i+8]=hi[i]; }
  return r;
}

// B-fragment (32x16 f16, KxN) fed from N-major (transposed) storage:
// lane = column n, lane-group g picks K halves [g*8, g*8+8) and [16+g*8, 16+g*8+8)
__device__ __forceinline__ v16h load_bfrag(const _Float16* rowPtr, int g){
  v8h lo = *(const v8h*)(rowPtr + g*8);
  v8h hi = *(const v8h*)(rowPtr + 16 + g*8);
  v16h r;
  #pragma unroll
  for(int i=0;i<8;i++){ r[i]=lo[i]; r[i+8]=hi[i]; }
  return r;
}

__device__ __forceinline__ v8f wmma_f16(v16h a, v16h b, v8f c){
  return __builtin_amdgcn_wmma_f32_16x16x32_f16(false, a, false, b, (short)0, c, false, false);
}

// Core: 32 tokens (2 M-tiles) x (TPW*16) cols; every B-fragment feeds 2 WMMAs.
template<int KDIM, int TPW>
__device__ __forceinline__ void gemm_core2(const _Float16* __restrict__ X,
                                           const _Float16* __restrict__ WT,
                                           int tokenBase, int nBase, int lane,
                                           v8f acc[2][TPW]){
  #pragma unroll
  for(int mt=0;mt<2;mt++)
    #pragma unroll
    for(int t=0;t<TPW;t++) acc[mt][t]=zero8();
  const int ml = lane & 15, g = lane >> 4;
  #pragma unroll
  for(int kk=0; kk<KDIM/32; ++kk){
    v16h a0 = load_afrag(X + (size_t)(tokenBase     + ml)*KDIM + kk*32, g);
    v16h a1 = load_afrag(X + (size_t)(tokenBase+16  + ml)*KDIM + kk*32, g);
    #pragma unroll
    for(int t=0;t<TPW;t++){
      v16h b = load_bfrag(WT + (size_t)(nBase + t*16 + ml)*KDIM + kk*32, g);
      acc[0][t] = wmma_f16(a0, b, acc[0][t]);
      acc[1][t] = wmma_f16(a1, b, acc[1][t]);
    }
  }
}

// ---------------------------------------------------------------------------
// Weight transpose + f16 convert:  W[k][cout] (f32)  ->  WT[cout][k] (f16)
// ---------------------------------------------------------------------------
__global__ void k_convert_wt(const float* __restrict__ W, _Float16* __restrict__ WT,
                             int K, int Cout){
  int idx = blockIdx.x*blockDim.x + threadIdx.x;
  if (idx >= K*Cout) return;
  int c = idx / K, k = idx - c*K;
  WT[idx] = (_Float16)W[(size_t)k*Cout + c];
}

// ---------------------------------------------------------------------------
// Per-token prep: anA = LN(A_I), sn = LN(S_I), sf = f16(S_I). one wave/token.
// ---------------------------------------------------------------------------
__global__ void k_token_prep(const float* __restrict__ A, const float* __restrict__ S,
                             _Float16* __restrict__ anA, _Float16* __restrict__ sn,
                             _Float16* __restrict__ sf){
  int lane = threadIdx.x & 31, wave = threadIdx.x >> 5;
  int token = blockIdx.x*4 + wave;
  const float4 av = *(const float4*)(A + (size_t)token*CDIM + lane*4);
  const float4 sv = *(const float4*)(S + (size_t)token*CDIM + lane*4);

  float s = av.x+av.y+av.z+av.w;
  float q = av.x*av.x+av.y*av.y+av.z*av.z+av.w*av.w;
  #pragma unroll
  for(int m=1;m<32;m<<=1){ s += __shfl_xor(s,m); q += __shfl_xor(q,m); }
  float mean = s*(1.0f/128.0f);
  float rstd = rsqrtf(q*(1.0f/128.0f) - mean*mean + 1e-5f);
  size_t o = (size_t)token*CDIM + lane*4;
  anA[o+0]=(_Float16)((av.x-mean)*rstd); anA[o+1]=(_Float16)((av.y-mean)*rstd);
  anA[o+2]=(_Float16)((av.z-mean)*rstd); anA[o+3]=(_Float16)((av.w-mean)*rstd);

  s = sv.x+sv.y+sv.z+sv.w;
  q = sv.x*sv.x+sv.y*sv.y+sv.z*sv.z+sv.w*sv.w;
  #pragma unroll
  for(int m=1;m<32;m<<=1){ s += __shfl_xor(s,m); q += __shfl_xor(q,m); }
  mean = s*(1.0f/128.0f);
  rstd = rsqrtf(q*(1.0f/128.0f) - mean*mean + 1e-5f);
  sn[o+0]=(_Float16)((sv.x-mean)*rstd); sn[o+1]=(_Float16)((sv.y-mean)*rstd);
  sn[o+2]=(_Float16)((sv.z-mean)*rstd); sn[o+3]=(_Float16)((sv.w-mean)*rstd);
  sf[o+0]=(_Float16)sv.x; sf[o+1]=(_Float16)sv.y;
  sf[o+2]=(_Float16)sv.z; sf[o+3]=(_Float16)sv.w;
}

// ---------------------------------------------------------------------------
// Pair bias at gathered (indicesQ, indicesK) pairs only (16x less Z traffic).
// LN over CP=16, project CP->H, fold -1e9*mask.   biasM[n][h][i][j]
// ---------------------------------------------------------------------------
__global__ void k_bias(const float* __restrict__ Z, const int* __restrict__ idxQ,
                       const int* __restrict__ idxK, const float* __restrict__ maskQ,
                       const float* __restrict__ maskK, const float* __restrict__ ln0g,
                       const float* __restrict__ ln0b, const float* __restrict__ Wb,
                       float* __restrict__ biasM){
  int e = blockIdx.x*blockDim.x + threadIdx.x;
  if (e >= NQBLK*QB_*KB_) return;
  int n = e >> 12, i = (e >> 7) & 31, j = e & 127;
  int qi = idxQ[n*QB_+i];
  int kj = idxK[n*KB_+j];
  const float* z = Z + ((size_t)qi*LSEQ + kj)*16;
  float v[16], s=0.f, q=0.f;
  #pragma unroll
  for(int c=0;c<16;c++){ v[c]=z[c]; s+=v[c]; q+=v[c]*v[c]; }
  float mean = s*(1.0f/16.0f);
  float rstd = rsqrtf(q*(1.0f/16.0f) - mean*mean + 1e-5f);
  float hb[4] = {0.f,0.f,0.f,0.f};
  #pragma unroll
  for(int c=0;c<16;c++){
    float zn = (v[c]-mean)*rstd*ln0g[c] + ln0b[c];
    #pragma unroll
    for(int h=0;h<4;h++) hb[h] += zn * Wb[c*4+h];
  }
  float mv = maskQ[n*QB_+i] + maskK[n*KB_+j];
  #pragma unroll
  for(int h=0;h<4;h++)
    biasM[(((size_t)n*NHEAD+h)*QB_+i)*KB_+j] = hb[h] - 1e9f*mv;
}

// ---------------------------------------------------------------------------
// adaLN GEMM: out = sigmoid(X@Wg+bg)*anIn + X@Ws.  WT=[Wg^T;Ws^T].
// Each wave computes BOTH halves for its 32-col slice -> pure register combine.
// ---------------------------------------------------------------------------
__global__ void k_gemm_adaln(const _Float16* __restrict__ X, const _Float16* __restrict__ WT,
                             const float* __restrict__ bg, const _Float16* __restrict__ anIn,
                             _Float16* __restrict__ out){
  int lane = threadIdx.x & 31, wave = threadIdx.x >> 5;
  int tokenBase = blockIdx.x*32, cBase = wave*32;
  const int ml = lane & 15, g = lane >> 4;
  v8f accG[2][2], accS[2][2];
  #pragma unroll
  for(int mt=0;mt<2;mt++)
    #pragma unroll
    for(int t=0;t<2;t++){ accG[mt][t]=zero8(); accS[mt][t]=zero8(); }
  #pragma unroll
  for(int kk=0;kk<4;kk++){
    v16h a0 = load_afrag(X + (size_t)(tokenBase    + ml)*CDIM + kk*32, g);
    v16h a1 = load_afrag(X + (size_t)(tokenBase+16 + ml)*CDIM + kk*32, g);
    #pragma unroll
    for(int t=0;t<2;t++){
      v16h bG = load_bfrag(WT + (size_t)(      cBase + t*16 + ml)*CDIM + kk*32, g);
      accG[0][t] = wmma_f16(a0, bG, accG[0][t]);
      accG[1][t] = wmma_f16(a1, bG, accG[1][t]);
      v16h bS = load_bfrag(WT + (size_t)(128 + cBase + t*16 + ml)*CDIM + kk*32, g);
      accS[0][t] = wmma_f16(a0, bS, accS[0][t]);
      accS[1][t] = wmma_f16(a1, bS, accS[1][t]);
    }
  }
  #pragma unroll
  for(int mt=0;mt<2;mt++)
    #pragma unroll
    for(int t=0;t<2;t++){
      int col = cBase + t*16 + ml;
      #pragma unroll
      for(int r=0;r<8;r++){
        int token = tokenBase + mt*16 + (g<<3) + r;
        size_t o = (size_t)token*CDIM + col;
        float gate = sigm(accG[mt][t][r] + bg[col]);
        out[o] = (_Float16)(gate*(float)anIn[o] + accS[mt][t][r]);
      }
    }
}

// ---------------------------------------------------------------------------
// QKVG GEMM: Cout=512 (Wq|Wk|Wv|Wg_attn). Q gets +bq, G gets sigmoid.
// ---------------------------------------------------------------------------
__global__ void k_gemm_qkvg(const _Float16* __restrict__ X, const _Float16* __restrict__ WT,
                            const float* __restrict__ bq, _Float16* __restrict__ Qf,
                            _Float16* __restrict__ Kf, _Float16* __restrict__ Vf,
                            _Float16* __restrict__ Gf){
  int lane = threadIdx.x & 31, wave = threadIdx.x >> 5;
  int tokenBase = blockIdx.x*32, nBase = wave*128;
  v8f acc[2][8];
  gemm_core2<128,8>(X, WT, tokenBase, nBase, lane, acc);
  const int ml = lane & 15, g = lane >> 4;
  #pragma unroll
  for(int mt=0;mt<2;mt++)
    #pragma unroll
    for(int t=0;t<8;t++){
      int col = nBase + t*16 + ml;
      int cg = col >> 7, c = col & 127;
      #pragma unroll
      for(int r=0;r<8;r++){
        int token = tokenBase + mt*16 + (g<<3) + r;
        size_t o = (size_t)token*CDIM + c;
        float v = acc[mt][t][r];
        if      (cg==0) Qf[o] = (_Float16)(v + bq[c]);
        else if (cg==1) Kf[o] = (_Float16)v;
        else if (cg==2) Vf[o] = (_Float16)v;
        else            Gf[o] = (_Float16)sigm(v);
      }
    }
}

// ---------------------------------------------------------------------------
// Gates GEMM: gate1=sigmoid(S@Wop+bop), gate2=sigmoid(S@ct_Wg+ct_bg)
// ---------------------------------------------------------------------------
__global__ void k_gemm_gates(const _Float16* __restrict__ X, const _Float16* __restrict__ WT,
                             const float* __restrict__ bop, const float* __restrict__ ctbg,
                             _Float16* __restrict__ gate1, _Float16* __restrict__ gate2){
  int lane = threadIdx.x & 31, wave = threadIdx.x >> 5;
  int tokenBase = blockIdx.x*32, nBase = wave*64;
  v8f acc[2][4];
  gemm_core2<128,4>(X, WT, tokenBase, nBase, lane, acc);
  const int ml = lane & 15, g = lane >> 4;
  #pragma unroll
  for(int mt=0;mt<2;mt++)
    #pragma unroll
    for(int t=0;t<4;t++){
      int col = nBase + t*16 + ml;
      #pragma unroll
      for(int r=0;r<8;r++){
        int token = tokenBase + mt*16 + (g<<3) + r;
        float v = acc[mt][t][r];
        if (col < 128) gate1[(size_t)token*CDIM + col]       = (_Float16)sigm(v + bop[col]);
        else           gate2[(size_t)token*CDIM + col - 128] = (_Float16)sigm(v + ctbg[col-128]);
      }
    }
}

// ---------------------------------------------------------------------------
// Windowed attention. One block per (n, d). 4 waves.
// Bias tile staged to LDS with GLOBAL_LOAD_ASYNC_TO_LDS_B128 (ASYNCcnt path).
// ---------------------------------------------------------------------------
__global__ void k_attn(const _Float16* __restrict__ Qf, const _Float16* __restrict__ Kf,
                       const _Float16* __restrict__ Vf, const _Float16* __restrict__ Gf,
                       const float* __restrict__ biasM, const int* __restrict__ idxQ,
                       const int* __restrict__ idxK, const float* __restrict__ maskQ,
                       _Float16* __restrict__ afg){
  __shared__ float sS[QB_][KB_];
  __shared__ __align__(16) float sBias[QB_][KB_];
  __shared__ __align__(32) _Float16 sP [QB_][KB_];
  __shared__ __align__(32) _Float16 sVT[HDIM][KB_];
  __shared__ int sKT[KB_];
  __shared__ int sQT[QB_];

  int tid = threadIdx.x, lane = tid & 31, wave = tid >> 5;
  int n = blockIdx.x, d = blockIdx.y;
  const size_t dbase = (size_t)d * LSEQ;

  if (tid < KB_) sKT[tid] = idxK[n*KB_ + tid];
  if (tid < QB_) sQT[tid] = idxQ[n*QB_ + tid];
  __syncthreads();

  const int ml = lane & 15, g = lane >> 4;
  const float scale = 0.17677669529663689f; // 1/sqrt(32)

  for(int h=0; h<NHEAD; ++h){
    // stage V^T : sVT[hd][j]
    for(int idx=tid; idx<HDIM*KB_; idx+=128){
      int hd = idx >> 7, j = idx & 127;
      sVT[hd][j] = Vf[(dbase + sKT[j])*CDIM + h*HDIM + hd];
    }
    // stage bias tile via async global->LDS (tracked by ASYNCcnt)
    {
      const float* bh = biasM + ((size_t)n*NHEAD + h)*QB_*KB_;
      unsigned ldsBase = (unsigned)(unsigned long long)(void*)&sBias[0][0];
      for(int idx=tid; idx<(QB_*KB_)/4; idx+=128){
        unsigned ldsAddr = ldsBase + (unsigned)idx*16u;
        const float* gp = bh + idx*4;
        asm volatile("global_load_async_to_lds_b128 %0, %1, off"
                     :: "v"(ldsAddr), "v"(gp) : "memory");
      }
#if __has_builtin(__builtin_amdgcn_s_wait_asynccnt)
      __builtin_amdgcn_s_wait_asynccnt(0);
#else
      asm volatile("s_wait_asynccnt 0x0" ::: "memory");
#endif
    }
    __syncthreads();

    // scores: wave handles jt in {2w,2w+1} x mt in {0,1}
    v16h aq[2];
    #pragma unroll
    for(int mt=0;mt<2;mt++)
      aq[mt] = load_afrag(Qf + (dbase + sQT[mt*16+ml])*CDIM + h*HDIM, g);

    #pragma unroll
    for(int jj=0;jj<2;jj++){
      int jt = wave*2 + jj;
      v16h bk = load_bfrag(Kf + (dbase + sKT[jt*16+ml])*CDIM + h*HDIM, g);
      #pragma unroll
      for(int mt=0;mt<2;mt++){
        v8f sc = wmma_f16(aq[mt], bk, zero8());
        int j = jt*16 + ml;
        #pragma unroll
        for(int r=0;r<8;r++){
          int i = mt*16 + (g<<3) + r;
          sS[i][j] = sc[r]*scale + sBias[i][j];
        }
      }
    }
    __syncthreads();

    // softmax over keys: 4 threads per row, 32 cols each
    {
      int i = tid >> 2, sg = tid & 3;
      float v[32], mx = -3.0e38f;
      #pragma unroll
      for(int u=0;u<32;u++){ v[u] = sS[i][sg*32+u]; mx = fmaxf(mx, v[u]); }
      #pragma unroll
      for(int m=1;m<4;m<<=1) mx = fmaxf(mx, __shfl_xor(mx,m));
      float sum = 0.f;
      #pragma unroll
      for(int u=0;u<32;u++){ v[u] = __expf(v[u]-mx); sum += v[u]; }
      #pragma unroll
      for(int m=1;m<4;m<<=1) sum += __shfl_xor(sum,m);
      float inv = 1.f/sum;
      #pragma unroll
      for(int u=0;u<32;u++) sP[i][sg*32+u] = (_Float16)(v[u]*inv);
    }
    __syncthreads();

    // O = P @ V^T-layout : each wave one 16x16 tile (mt = w>>1, nt = w&1)
    {
      int mt = wave >> 1, nt = wave & 1;
      v8f o = zero8();
      #pragma unroll
      for(int kk=0;kk<4;kk++){
        v16h a = load_afrag(&sP [mt*16 + ml][kk*32], g);
        v16h b = load_bfrag(&sVT[nt*16 + ml][kk*32], g);
        o = wmma_f16(a, b, o);
      }
      int hd  = nt*16 + ml;
      int col = h*HDIM + hd;
      #pragma unroll
      for(int r=0;r<8;r++){
        int i  = mt*16 + (g<<3) + r;
        int qi = sQT[i];
        float gg = (float)Gf[(dbase + qi)*CDIM + col];
        float mq = maskQ[n*QB_ + i];
        afg[(dbase + qi)*CDIM + col] = (_Float16)(o[r]*gg*(1.f-mq));
      }
    }
    __syncthreads();
  }
}

// ---------------------------------------------------------------------------
// Wa GEMM + first residual + LN(A):  A = A_I + gate1*(afg@Wa); lnA = LN(A)
// ---------------------------------------------------------------------------
__global__ void k_gemm_wa(const _Float16* __restrict__ X, const _Float16* __restrict__ WT,
                          const float* __restrict__ A_I, const _Float16* __restrict__ gate1,
                          float* __restrict__ Abuf, _Float16* __restrict__ lnA){
  __shared__ float sA[32][128];
  int lane = threadIdx.x & 31, wave = threadIdx.x >> 5;
  int tokenBase = blockIdx.x*32, nBase = wave*32;
  v8f acc[2][2];
  gemm_core2<128,2>(X, WT, tokenBase, nBase, lane, acc);
  const int ml = lane & 15, g = lane >> 4;
  #pragma unroll
  for(int mt=0;mt<2;mt++)
    #pragma unroll
    for(int t=0;t<2;t++){
      int col = nBase + t*16 + ml;
      #pragma unroll
      for(int r=0;r<8;r++){
        int m = mt*16 + (g<<3) + r;
        size_t o = (size_t)(tokenBase+m)*CDIM + col;
        float a = A_I[o] + (float)gate1[o]*acc[mt][t][r];
        Abuf[o] = a;
        sA[m][col] = a;
      }
    }
  __syncthreads();
  // row LN: 4 threads per token (32 tokens, 128 threads)
  int tt = threadIdx.x >> 2, sg = threadIdx.x & 3;
  float vals[32], s=0.f, q=0.f;
  #pragma unroll
  for(int u=0;u<32;u++){ float x = sA[tt][sg*32+u]; vals[u]=x; s+=x; q+=x*x; }
  #pragma unroll
  for(int m=1;m<4;m<<=1){ s += __shfl_xor(s,m); q += __shfl_xor(q,m); }
  float mean = s*(1.0f/128.0f);
  float rstd = rsqrtf(q*(1.0f/128.0f) - mean*mean + 1e-5f);
  #pragma unroll
  for(int u=0;u<32;u++)
    lnA[(size_t)(tokenBase+tt)*CDIM + sg*32+u] = (_Float16)((vals[u]-mean)*rstd);
}

// ---------------------------------------------------------------------------
// SwiGLU GEMM: WT=[W_sw^T;W_lin^T]; each wave computes both halves of its
// 64-col slice -> register combine, direct f16 store.  b = silu(sw)*lin
// ---------------------------------------------------------------------------
__global__ void k_gemm_swlin(const _Float16* __restrict__ X, const _Float16* __restrict__ WT,
                             _Float16* __restrict__ bbuf){
  int lane = threadIdx.x & 31, wave = threadIdx.x >> 5;
  int tokenBase = blockIdx.x*32, cBase = wave*64;
  const int ml = lane & 15, g = lane >> 4;
  v8f accW[2][4], accL[2][4];
  #pragma unroll
  for(int mt=0;mt<2;mt++)
    #pragma unroll
    for(int t=0;t<4;t++){ accW[mt][t]=zero8(); accL[mt][t]=zero8(); }
  #pragma unroll
  for(int kk=0;kk<4;kk++){
    v16h a0 = load_afrag(X + (size_t)(tokenBase    + ml)*CDIM + kk*32, g);
    v16h a1 = load_afrag(X + (size_t)(tokenBase+16 + ml)*CDIM + kk*32, g);
    #pragma unroll
    for(int t=0;t<4;t++){
      v16h bW = load_bfrag(WT + (size_t)(      cBase + t*16 + ml)*CDIM + kk*32, g);
      accW[0][t] = wmma_f16(a0, bW, accW[0][t]);
      accW[1][t] = wmma_f16(a1, bW, accW[1][t]);
      v16h bL = load_bfrag(WT + (size_t)(256 + cBase + t*16 + ml)*CDIM + kk*32, g);
      accL[0][t] = wmma_f16(a0, bL, accL[0][t]);
      accL[1][t] = wmma_f16(a1, bL, accL[1][t]);
    }
  }
  #pragma unroll
  for(int mt=0;mt<2;mt++)
    #pragma unroll
    for(int t=0;t<4;t++){
      int col = cBase + t*16 + ml;
      #pragma unroll
      for(int r=0;r<8;r++){
        int token = tokenBase + mt*16 + (g<<3) + r;
        float x = accW[mt][t][r], y = accL[mt][t][r];
        bbuf[(size_t)token*256 + col] = (_Float16)(x*sigm(x)*y);
      }
    }
}

// ---------------------------------------------------------------------------
// Final GEMM: out = Abuf + gate2 * (b @ W_out2)   (K=256, Cout=128, fp32 out)
// ---------------------------------------------------------------------------
__global__ void k_gemm_out2(const _Float16* __restrict__ X, const _Float16* __restrict__ WT,
                            const float* __restrict__ Abuf, const _Float16* __restrict__ gate2,
                            float* __restrict__ out){
  int lane = threadIdx.x & 31, wave = threadIdx.x >> 5;
  int tokenBase = blockIdx.x*32, nBase = wave*32;
  v8f acc[2][2];
  gemm_core2<256,2>(X, WT, tokenBase, nBase, lane, acc);
  const int ml = lane & 15, g = lane >> 4;
  #pragma unroll
  for(int mt=0;mt<2;mt++)
    #pragma unroll
    for(int t=0;t<2;t++){
      int col = nBase + t*16 + ml;
      #pragma unroll
      for(int r=0;r<8;r++){
        int token = tokenBase + mt*16 + (g<<3) + r;
        size_t o = (size_t)token*CDIM + col;
        out[o] = Abuf[o] + (float)gate2[o]*acc[mt][t][r];
      }
    }
}

// ---------------------------------------------------------------------------
extern "C" void kernel_launch(void* const* d_in, const int* in_sizes, int n_in,
                              void* d_out, int out_size, void* d_ws, size_t ws_size,
                              hipStream_t stream){
  (void)in_sizes; (void)n_in; (void)out_size; (void)ws_size;
  const float* A_I     = (const float*)d_in[0];
  const float* S_I     = (const float*)d_in[1];
  const float* Z       = (const float*)d_in[2];
  const int*   idxQ    = (const int*)d_in[3];
  const int*   idxK    = (const int*)d_in[4];
  const float* maskQ   = (const float*)d_in[5];
  const float* maskK   = (const float*)d_in[6];
  const float* ada1_Wg = (const float*)d_in[7];
  const float* ada1_bg = (const float*)d_in[8];
  const float* ada1_Ws = (const float*)d_in[9];
  const float* Wq      = (const float*)d_in[10];
  const float* bq      = (const float*)d_in[11];
  const float* Wk      = (const float*)d_in[12];
  const float* Wv      = (const float*)d_in[13];
  const float* ln0g    = (const float*)d_in[14];
  const float* ln0b    = (const float*)d_in[15];
  const float* Wb      = (const float*)d_in[16];
  const float* Wg_attn = (const float*)d_in[17];
  const float* Wa      = (const float*)d_in[18];
  const float* Wop     = (const float*)d_in[19];
  const float* bop     = (const float*)d_in[20];
  const float* ada2_Wg = (const float*)d_in[21];
  const float* ada2_bg = (const float*)d_in[22];
  const float* ada2_Ws = (const float*)d_in[23];
  const float* W_sw    = (const float*)d_in[24];
  const float* W_lin   = (const float*)d_in[25];
  const float* ct_Wg   = (const float*)d_in[26];
  const float* ct_bg   = (const float*)d_in[27];
  const float* W_out2  = (const float*)d_in[28];
  float* out = (float*)d_out;

  char* w = (char*)d_ws;
  size_t off = 0;
  auto alloc = [&](size_t b)->char*{ char* p = w + off; off += (b + 255) & ~(size_t)255; return p; };

  _Float16* wt_ada1  = (_Float16*)alloc(256*128*2);
  _Float16* wt_qkvg  = (_Float16*)alloc(512*128*2);
  _Float16* wt_gates = (_Float16*)alloc(256*128*2);
  _Float16* wt_wa    = (_Float16*)alloc(128*128*2);
  _Float16* wt_ada2  = (_Float16*)alloc(256*128*2);
  _Float16* wt_swlin = (_Float16*)alloc(512*128*2);
  _Float16* wt_out2  = (_Float16*)alloc(128*256*2);
  const size_t TB = (size_t)NTOK*CDIM;
  _Float16* anA   = (_Float16*)alloc(TB*2);
  _Float16* snb   = (_Float16*)alloc(TB*2);
  _Float16* sfb   = (_Float16*)alloc(TB*2);
  _Float16* Anb   = (_Float16*)alloc(TB*2);
  _Float16* Qf    = (_Float16*)alloc(TB*2);
  _Float16* Kf    = (_Float16*)alloc(TB*2);
  _Float16* Vf    = (_Float16*)alloc(TB*2);
  _Float16* Gf    = (_Float16*)alloc(TB*2);
  float*    biasM = (float*)   alloc((size_t)NQBLK*NHEAD*QB_*KB_*4);
  _Float16* afg   = (_Float16*)alloc(TB*2);
  _Float16* gate1 = (_Float16*)alloc(TB*2);
  _Float16* gate2 = (_Float16*)alloc(TB*2);
  float*    Abuf  = (float*)   alloc(TB*4);
  _Float16* lnA   = (_Float16*)alloc(TB*2);
  _Float16* an2   = (_Float16*)alloc(TB*2);
  _Float16* bbuf  = (_Float16*)alloc((size_t)NTOK*256*2);

  auto conv = [&](const float* W, _Float16* WT, int K, int Cout){
    int total = K*Cout;
    k_convert_wt<<<(total+255)/256, 256, 0, stream>>>(W, WT, K, Cout);
  };
  conv(ada1_Wg, wt_ada1,           128, 128);
  conv(ada1_Ws, wt_ada1 + 16384,   128, 128);
  conv(Wq,      wt_qkvg,           128, 128);
  conv(Wk,      wt_qkvg + 16384,   128, 128);
  conv(Wv,      wt_qkvg + 32768,   128, 128);
  conv(Wg_attn, wt_qkvg + 49152,   128, 128);
  conv(Wop,     wt_gates,          128, 128);
  conv(ct_Wg,   wt_gates + 16384,  128, 128);
  conv(Wa,      wt_wa,             128, 128);
  conv(ada2_Wg, wt_ada2,           128, 128);
  conv(ada2_Ws, wt_ada2 + 16384,   128, 128);
  conv(W_sw,    wt_swlin,          128, 256);
  conv(W_lin,   wt_swlin + 32768,  128, 256);
  conv(W_out2,  wt_out2,           256, 128);

  k_token_prep<<<NTOK/4, 128, 0, stream>>>(A_I, S_I, anA, snb, sfb);
  k_bias<<<(NQBLK*QB_*KB_)/256, 256, 0, stream>>>(Z, idxQ, idxK, maskQ, maskK,
                                                  ln0g, ln0b, Wb, biasM);

  dim3 gb(NTOK/32);
  k_gemm_adaln<<<gb, 128, 0, stream>>>(snb, wt_ada1, ada1_bg, anA, Anb);
  k_gemm_qkvg <<<gb, 128, 0, stream>>>(Anb, wt_qkvg, bq, Qf, Kf, Vf, Gf);
  k_gemm_gates<<<gb, 128, 0, stream>>>(sfb, wt_gates, bop, ct_bg, gate1, gate2);

  k_attn<<<dim3(NQBLK, 8), 128, 0, stream>>>(Qf, Kf, Vf, Gf, biasM, idxQ, idxK, maskQ, afg);

  k_gemm_wa   <<<gb, 128, 0, stream>>>(afg, wt_wa, A_I, gate1, Abuf, lnA);
  k_gemm_adaln<<<gb, 128, 0, stream>>>(snb, wt_ada2, ada2_bg, lnA, an2);
  k_gemm_swlin<<<gb, 128, 0, stream>>>(an2, wt_swlin, bbuf);
  k_gemm_out2 <<<gb, 128, 0, stream>>>(bbuf, wt_out2, Abuf, gate2, out);
}
